// scaled_dot_attention_13683765805757
// MI455X (gfx1250) — compile-verified, matrix operand order checked by reading, not run on hardware
//
#include <hip/hip_runtime.h>
#include <stdint.h>

typedef __attribute__((ext_vector_type(16))) _Float16 v16h;
typedef __attribute__((ext_vector_type(8)))  _Float16 v8h;
typedef __attribute__((ext_vector_type(8)))  float    v8f;
typedef __attribute__((ext_vector_type(4)))  float    v4f;
typedef __attribute__((ext_vector_type(2)))  uint32_t v2u;

#define S_LEN 2048
#define D_DIM 64
#define NBATCH 32
#define WAVES 8
#define QTILE 128   // queries per block (8 waves x 16 rows)
#define KTILE 32    // keys per iteration

// cheap 2-mul integer hash; 4 random bits per element (keep if nibble < 13, p~0.8125).
// (jax threefry is not replicable on-device; statistical equivalent.)
__device__ __forceinline__ uint32_t hash32(uint32_t x) {
    x *= 0x9E3779B9u;
    x ^= x >> 16;
    x *= 0x85ebca6bu;
    x ^= x >> 13;
    return x;
}

// pack two contiguous 8-half LDS chunks into one v16h operand
__device__ __forceinline__ v16h lds_pack(const _Float16* c0, const _Float16* c1) {
    v8h lo = *(const v8h*)c0;
    v8h hi = *(const v8h*)c1;
    return __builtin_shufflevector(lo, hi, 0,1,2,3,4,5,6,7,8,9,10,11,12,13,14,15);
}

__global__ __launch_bounds__(256)
void fa_fwd_kernel(const float* __restrict__ K, const float* __restrict__ V,
                   const float* __restrict__ Q, const unsigned char* __restrict__ mask,
                   float* __restrict__ out)
{
    __shared__ alignas(16) _Float16 sK [KTILE * D_DIM];   // [key][d] row-major (A for S^T)
    __shared__ alignas(16) _Float16 sVt[D_DIM * KTILE];   // [d][key] transposed  (A for O^T)

    const int b    = blockIdx.y;
    const int q0   = blockIdx.x * QTILE;
    const int tid  = threadIdx.x;
    const int wave = tid >> 5;
    const int lane = tid & 31;
    const int l16  = lane & 15;
    const int hi8  = (lane >= 16) ? 8 : 0;
    const int cb   = hi8;                    // K-chunk base for 16-bit A/B packing

    const int q = q0 + wave * 16 + l16;      // this lane's query row (pairs of lanes share q)

    // ---- B operand: Q^T (d x q), lane holds column q, d-chunks {cb..cb+7, cb+16..cb+23}.
    //      Fold in 1/sqrt(D) * 1/(1-p) = 1/(8*0.8).
    const float qscale = 1.0f / (8.0f * 0.8f);
    v16h bq0, bq1;
    {
        const float* qp = Q + ((size_t)b * S_LEN + q) * D_DIM;
#pragma unroll
        for (int i = 0; i < 8; ++i) {
            bq0[i]     = (_Float16)(qp[cb + i]           * qscale);
            bq0[8 + i] = (_Float16)(qp[cb + 16 + i]      * qscale);
            bq1[i]     = (_Float16)(qp[32 + cb + i]      * qscale);
            bq1[8 + i] = (_Float16)(qp[32 + cb + 16 + i] * qscale);
        }
    }

    // O^T accumulator: 4 tiles (d 0-15,16-31,32-47,48-63); lane = q, reg r = d_local
    v8f o0 = {}, o1 = {}, o2 = {}, o3 = {};
    float m = -1e30f, l = 0.0f;

    const unsigned char* mrow = mask + (size_t)((uint32_t)(b * S_LEN + q)) * S_LEN + hi8;
    const uint32_t flbase = (uint32_t)(b * S_LEN + q) * (uint32_t)S_LEN + (uint32_t)hi8;

    const int nTiles = S_LEN / KTILE;
    for (int kt = 0; kt < nTiles; ++kt) {
        const int kbase = kt * KTILE;

        // ---- cooperative staging: 2048 f16 elems each of K and V, 8 per thread
        {
            const int flat = tid * 8;
            const int key  = flat >> 6;      // /64
            const int d    = flat & 63;
            const float* kp = K + ((size_t)b * S_LEN + kbase + key) * D_DIM + d;
            const float* vp = V + ((size_t)b * S_LEN + kbase + key) * D_DIM + d;
            _Float16* dk = &sK[key * D_DIM + d];
#pragma unroll
            for (int i = 0; i < 8; ++i) dk[i] = (_Float16)kp[i];
#pragma unroll
            for (int i = 0; i < 8; ++i) sVt[(d + i) * KTILE + key] = (_Float16)vp[i];
            if (kt + 1 < nTiles)
                __builtin_prefetch(kp + KTILE * D_DIM, 0, 1);   // global_prefetch_b8
        }
        __syncthreads();

        // ---- S^T = K x Q^T : two 16(k) x 16(q) tiles, K-dim 64 = 2 WMMAs each.
        //      A = K rows (lane = key row), B = Q^T (in registers).
        v8f s0 = {}, s1 = {};
        {
            const _Float16* kr0 = &sK[l16 * D_DIM];          // key rows 0-15
            const _Float16* kr1 = &sK[(16 + l16) * D_DIM];   // key rows 16-31
            v16h ak;
            ak = lds_pack(kr0 + cb,      kr0 + cb + 16);                 // d 0-31
            s0 = __builtin_amdgcn_wmma_f32_16x16x32_f16(false, ak, false, bq0, (short)0, s0, false, false);
            ak = lds_pack(kr0 + 32 + cb, kr0 + 32 + cb + 16);            // d 32-63
            s0 = __builtin_amdgcn_wmma_f32_16x16x32_f16(false, ak, false, bq1, (short)0, s0, false, false);
            ak = lds_pack(kr1 + cb,      kr1 + cb + 16);
            s1 = __builtin_amdgcn_wmma_f32_16x16x32_f16(false, ak, false, bq0, (short)0, s1, false, false);
            ak = lds_pack(kr1 + 32 + cb, kr1 + 32 + cb + 16);
            s1 = __builtin_amdgcn_wmma_f32_16x16x32_f16(false, ak, false, bq1, (short)0, s1, false, false);
        }

        // ---- mask + dropout + online softmax, all in-lane (lane = fixed q row).
        //      This lane's elements: k = kbase + hi8 + r  (tile0) and +16 (tile1).
        const v2u mw0 = __builtin_nontemporal_load((const v2u*)(mrow + kbase));
        const v2u mw1 = __builtin_nontemporal_load((const v2u*)(mrow + kbase + 16));
        const uint32_t fl0 = flbase + (uint32_t)kbase;
        const uint32_t h0 = hash32(fl0);         // 8 nibbles -> tile0 drop decisions
        const uint32_t h1 = hash32(fl0 + 16);    // 8 nibbles -> tile1 drop decisions

        float rmax = -1e30f;
#pragma unroll
        for (int r = 0; r < 8; ++r) {
            float x0 = s0[r];
            float x1 = s1[r];
            // mask byte LSB is the flag (bool is 0/1): x += bit * -3e38  (acts as -inf)
            const uint32_t b0 = (mw0[r >> 2] >> (8 * (r & 3))) & 1u;
            const uint32_t b1 = (mw1[r >> 2] >> (8 * (r & 3))) & 1u;
            x0 = fmaf((float)b0, -3.0e38f, x0);
            x1 = fmaf((float)b1, -3.0e38f, x1);
            // dropout: 4 random bits per element
            if (((h0 >> (4 * r)) & 0xFu) >= 13u) x0 = 0.0f;
            if (((h1 >> (4 * r)) & 0xFu) >= 13u) x1 = 0.0f;
            s0[r] = x0; s1[r] = x1;
            rmax = fmaxf(rmax, fmaxf(x0, x1));
        }
        rmax = fmaxf(rmax, __shfl_xor(rmax, 16));   // merge the two k-halves of this q row

        const float mnew  = fmaxf(m, rmax);
        const float alpha = __expf(m - mnew);
        float rs = 0.0f;
        v16h bP;                                    // P^T B-operand, pure register repack
#pragma unroll
        for (int r = 0; r < 8; ++r) {
            const float p0 = __expf(s0[r] - mnew);
            const float p1 = __expf(s1[r] - mnew);
            rs += p0 + p1;
            bP[r]     = (_Float16)p0;               // k chunk {cb..cb+7}
            bP[8 + r] = (_Float16)p1;               // k chunk {cb+16..cb+23}
        }
        rs += __shfl_xor(rs, 16);
        l = l * alpha + rs;
        m = mnew;

#pragma unroll
        for (int r = 0; r < 8; ++r) { o0[r] *= alpha; o1[r] *= alpha; o2[r] *= alpha; o3[r] *= alpha; }

        // ---- O^T += V^T(16d x 32k) x P^T(32k x 16q): A from sVt, B = bP (shared by all 4)
        {
            const _Float16* v0 = &sVt[(0 * 16 + l16) * KTILE];
            const _Float16* v1 = &sVt[(1 * 16 + l16) * KTILE];
            const _Float16* v2 = &sVt[(2 * 16 + l16) * KTILE];
            const _Float16* v3 = &sVt[(3 * 16 + l16) * KTILE];
            o0 = __builtin_amdgcn_wmma_f32_16x16x32_f16(false, lds_pack(v0 + cb, v0 + cb + 16), false, bP, (short)0, o0, false, false);
            o1 = __builtin_amdgcn_wmma_f32_16x16x32_f16(false, lds_pack(v1 + cb, v1 + cb + 16), false, bP, (short)0, o1, false, false);
            o2 = __builtin_amdgcn_wmma_f32_16x16x32_f16(false, lds_pack(v2 + cb, v2 + cb + 16), false, bP, (short)0, o2, false, false);
            o3 = __builtin_amdgcn_wmma_f32_16x16x32_f16(false, lds_pack(v3 + cb, v3 + cb + 16), false, bP, (short)0, o3, false, false);
        }
        __syncthreads();   // protect sK/sVt before next iteration's staging
    }

    // ---- normalize and store: lane holds output row q, regs hold d = t*16 + hi8 + r
    {
        const float inv = 1.0f / l;
        float* orow = out + ((size_t)b * S_LEN + q) * D_DIM + hi8;
        v4f w;
        w = (v4f){o0[0]*inv, o0[1]*inv, o0[2]*inv, o0[3]*inv};
        __builtin_nontemporal_store(w, (v4f*)(orow + 0));
        w = (v4f){o0[4]*inv, o0[5]*inv, o0[6]*inv, o0[7]*inv};
        __builtin_nontemporal_store(w, (v4f*)(orow + 4));
        w = (v4f){o1[0]*inv, o1[1]*inv, o1[2]*inv, o1[3]*inv};
        __builtin_nontemporal_store(w, (v4f*)(orow + 16));
        w = (v4f){o1[4]*inv, o1[5]*inv, o1[6]*inv, o1[7]*inv};
        __builtin_nontemporal_store(w, (v4f*)(orow + 20));
        w = (v4f){o2[0]*inv, o2[1]*inv, o2[2]*inv, o2[3]*inv};
        __builtin_nontemporal_store(w, (v4f*)(orow + 32));
        w = (v4f){o2[4]*inv, o2[5]*inv, o2[6]*inv, o2[7]*inv};
        __builtin_nontemporal_store(w, (v4f*)(orow + 36));
        w = (v4f){o3[0]*inv, o3[1]*inv, o3[2]*inv, o3[3]*inv};
        __builtin_nontemporal_store(w, (v4f*)(orow + 48));
        w = (v4f){o3[4]*inv, o3[5]*inv, o3[6]*inv, o3[7]*inv};
        __builtin_nontemporal_store(w, (v4f*)(orow + 52));
    }
}

extern "C" void kernel_launch(void* const* d_in, const int* in_sizes, int n_in,
                              void* d_out, int out_size, void* d_ws, size_t ws_size,
                              hipStream_t stream) {
    // setup_inputs order: K, V, Q, mask
    const float* K = (const float*)d_in[0];
    const float* V = (const float*)d_in[1];
    const float* Q = (const float*)d_in[2];
    const unsigned char* mask = (const unsigned char*)d_in[3];
    float* out = (float*)d_out;

    dim3 grid(S_LEN / QTILE, NBATCH);
    fa_fwd_kernel<<<grid, 256, 0, stream>>>(K, V, Q, mask, out);
}
